// HouseholderFlow_40492951667023
// MI455X (gfx1250) — matrix operand order repaired and checked by reading, not compiled, not run on hardware
//
#include <hip/hip_runtime.h>

// Problem constants (match the reference)
#define B_DIM 8192
#define L_DIM 2048
#define K_DIM 2048
#define NFLOW 8

#define KB    64          // K elements staged per pipeline stage (2 WMMA sub-steps)
#define LDP   72          // padded LDS row stride in bf16 elements (144B: bank-conflict-free)

typedef __attribute__((ext_vector_type(16))) __bf16    v16bf;
typedef __attribute__((ext_vector_type(8)))  float     v8f;
typedef __attribute__((ext_vector_type(4)))  unsigned  u4v;
typedef __attribute__((ext_vector_type(2)))  unsigned  u2v;
typedef __attribute__((ext_vector_type(4)))  float     f4v;

union Frag {
    v16bf v;
    u4v   q[2];
};

#if defined(__has_builtin)
#  if __has_builtin(__builtin_amdgcn_global_load_async_to_lds_b128)
#    define HAVE_ASYNC_LDS 1
#  endif
#endif

#if defined(HAVE_ASYNC_LDS)
// Builtin signature (from hipcc diagnostics): param0 = int4 addrspace(1)* (non-const),
// param1 = int4 addrspace(3)*, then constant-int offset and cpol.
typedef int v4i __attribute__((vector_size(16)));
typedef __attribute__((address_space(1))) v4i* gas_i4p;
typedef __attribute__((address_space(3))) v4i* las_i4p;
#endif

__device__ __forceinline__ void wait_async_lds() {
#if defined(HAVE_ASYNC_LDS)
#  if __has_builtin(__builtin_amdgcn_s_wait_asynccnt)
    __builtin_amdgcn_s_wait_asynccnt(0);
#  else
    asm volatile("s_wait_asynccnt 0" ::: "memory");
#  endif
#endif
}

__device__ __forceinline__ unsigned short f2bf(float f) {
    unsigned u = __float_as_uint(f);
    u += 0x7FFFu + ((u >> 16) & 1u);   // round-to-nearest-even
    return (unsigned short)(u >> 16);
}
__device__ __forceinline__ unsigned pack2(float a, float b) {
    return (unsigned)f2bf(a) | ((unsigned)f2bf(b) << 16);
}
__device__ __forceinline__ float bfhi2f(unsigned packed) {   // high bf16 of a dword
    return __uint_as_float(packed & 0xFFFF0000u);
}
__device__ __forceinline__ float bflo2f(unsigned packed) {   // low bf16 of a dword
    return __uint_as_float(packed << 16);
}

// ---------------------------------------------------------------------------
// Stage one 128xKB bf16 tile of A and of W into (padded) LDS.
// Pure byte copy: async-to-LDS when available, else batched b128 loads/stores.
// ---------------------------------------------------------------------------
__device__ __forceinline__ void stage_tiles(const unsigned short* __restrict__ Ab,
                                            const unsigned short* __restrict__ Wb,
                                            unsigned short* lA, unsigned short* lB,
                                            int bm, int bn, int kk, int t)
{
#if defined(HAVE_ASYNC_LDS)
    #pragma unroll
    for (int p = 0; p < 4; ++p) {
        const int c   = t + 256 * p;      // chunk id: 1024 x 16B per matrix
        const int row = c >> 3;           // 8 chunks per 64-elem row
        const int ko  = (c & 7) * 8;
        __builtin_amdgcn_global_load_async_to_lds_b128(
            (gas_i4p)(unsigned short*)&Ab[(size_t)(bm + row) * K_DIM + kk + ko],
            (las_i4p)&lA[row * LDP + ko], 0, 0);
        __builtin_amdgcn_global_load_async_to_lds_b128(
            (gas_i4p)(unsigned short*)&Wb[(size_t)(bn + row) * K_DIM + kk + ko],
            (las_i4p)&lB[row * LDP + ko], 0, 0);
    }
#else
    u4v ra[4], rb[4];
    int rows[4], kos[4];
    #pragma unroll
    for (int p = 0; p < 4; ++p) {         // batch the global loads first
        const int c = t + 256 * p;
        rows[p] = c >> 3;
        kos[p]  = (c & 7) * 8;
        ra[p] = *(const u4v*)&Ab[(size_t)(bm + rows[p]) * K_DIM + kk + kos[p]];
        rb[p] = *(const u4v*)&Wb[(size_t)(bn + rows[p]) * K_DIM + kk + kos[p]];
    }
    #pragma unroll
    for (int p = 0; p < 4; ++p) {
        *(u4v*)&lA[rows[p] * LDP + kos[p]] = ra[p];
        *(u4v*)&lB[rows[p] * LDP + kos[p]] = rb[p];
    }
#endif
}

// ---------------------------------------------------------------------------
// Fused bf16 GEMM + partial Householder dots.
//   out[b,n] (bf16) = sum_k A[b,k]*W[n,k] + bias[n]
//   vz[b] += sum_n out*z ; vv[b] += sum_n out^2   (atomic f32 partials)
// Block tile 128x128, 8 waves, wave tile 32x64 (2x4 WMMA 16x16),
// double-buffered async LDS pipeline, KB=64 per stage.
// ---------------------------------------------------------------------------
__global__ void __launch_bounds__(256)
hf_gemm(const unsigned short* __restrict__ A,      // [B, K] bf16 activations
        const unsigned short* __restrict__ W,      // [L, K] bf16 weights
        const float* __restrict__ bias,            // [L]
        const float* __restrict__ z,               // [B, L] current z (f32)
        unsigned short* __restrict__ out,          // [B, L] v_new (bf16)
        float* __restrict__ vz,                    // [B]
        float* __restrict__ vv)                    // [B]
{
    __shared__ unsigned short lsA[2][128 * LDP];
    __shared__ unsigned short lsB[2][128 * LDP];

    const int t    = threadIdx.x;
    const int lane = t & 31;
    const int w    = t >> 5;
    const int wm   = w >> 1;          // 0..3 -> 32-row group
    const int wn   = w & 1;           // 0..1 -> 64-col group
    const int nl   = lane & 15;
    const int half = lane >> 4;

    const int bm = blockIdx.y * 128;
    const int bn = blockIdx.x * 128;

    const v8f vzero = {0.f, 0.f, 0.f, 0.f, 0.f, 0.f, 0.f, 0.f};
    v8f acc[2][4];
    #pragma unroll
    for (int i = 0; i < 2; ++i)
        #pragma unroll
        for (int j = 0; j < 4; ++j) acc[i][j] = vzero;

    // prologue: stage first tile
    stage_tiles(A, W, lsA[0], lsB[0], bm, bn, 0, t);
    wait_async_lds();
    __syncthreads();

    int cur = 0;
    for (int kk = 0; kk < K_DIM; kk += KB) {
        if (kk + KB < K_DIM)   // stream next stage while computing this one
            stage_tiles(A, W, lsA[cur ^ 1], lsB[cur ^ 1], bm, bn, kk + KB, t);

        #pragma unroll
        for (int ks = 0; ks < 2; ++ks) {               // two K=32 sub-steps
            Frag af[2], bfr[4];
            #pragma unroll
            for (int im = 0; im < 2; ++im) {
                const int base = (32 * wm + 16 * im + nl) * LDP + ks * 32;
                af[im].q[0] = *(const u4v*)&lsA[cur][base + half * 8];
                af[im].q[1] = *(const u4v*)&lsA[cur][base + 16 + half * 8];
            }
            #pragma unroll
            for (int in = 0; in < 4; ++in) {
                const int base = (64 * wn + 16 * in + nl) * LDP + ks * 32 + half * 8;
                bfr[in].q[0] = *(const u4v*)&lsB[cur][base];
                bfr[in].q[1] = *(const u4v*)&lsB[cur][base + 16];
            }
            #pragma unroll
            for (int im = 0; im < 2; ++im)
                #pragma unroll
                for (int in = 0; in < 4; ++in)
                    acc[im][in] = __builtin_amdgcn_wmma_f32_16x16x32_bf16(
                        false, af[im].v, false, bfr[in].v,
                        (short)0, acc[im][in], false, false);
        }

        wait_async_lds();      // next buffer landed
        __syncthreads();       // everyone done reading cur / writing cur^1
        cur ^= 1;
    }

    // ---- epilogue: bias, bf16 store of v_new, fused partial dot products ----
    #pragma unroll
    for (int im = 0; im < 2; ++im) {
        float svz[8], svv[8];
        #pragma unroll
        for (int r = 0; r < 8; ++r) { svz[r] = 0.f; svv[r] = 0.f; }

        #pragma unroll
        for (int in = 0; in < 4; ++in) {
            const int gn = bn + 64 * wn + 16 * in + nl;
            const float bv = bias[gn];
            #pragma unroll
            for (int r = 0; r < 8; ++r) {
                const int gm = bm + 32 * wm + 16 * im + half * 8 + r;
                const float val = acc[im][in][r] + bv;
                out[(size_t)gm * L_DIM + gn] = f2bf(val);
                const float zval = z[(size_t)gm * L_DIM + gn];
                svz[r] += val * zval;
                svv[r] += val * val;
            }
        }
        #pragma unroll
        for (int r = 0; r < 8; ++r) {
            float a = svz[r], b = svv[r];
            #pragma unroll
            for (int m = 1; m < 16; m <<= 1) {   // xor masks stay within 16-lane half
                a += __shfl_xor(a, m, 32);
                b += __shfl_xor(b, m, 32);
            }
            if (nl == 0) {
                const int gm = bm + 32 * wm + 16 * im + half * 8 + r;
                atomicAdd(&vz[gm], a);
                atomicAdd(&vv[gm], b);
            }
        }
    }
}

// f32 -> bf16 bulk converter (4 elements / thread)
__global__ void __launch_bounds__(256)
cvt_bf16(const float* __restrict__ in, unsigned short* __restrict__ out, long long n4)
{
    const long long i = (long long)blockIdx.x * blockDim.x + threadIdx.x;
    if (i >= n4) return;
    const f4v a = ((const f4v*)in)[i];
    u2v p; p.x = pack2(a.x, a.y); p.y = pack2(a.z, a.w);
    ((u2v*)out)[i] = p;
}

// z[b,:] -= (2 * vz[b] / vv[b]) * v[b,:]   (v is bf16)
__global__ void __launch_bounds__(256)
hf_update(float* __restrict__ z, const unsigned short* __restrict__ v,
          const float* __restrict__ vz, const float* __restrict__ vv)
{
    const size_t idx = (size_t)blockIdx.x * blockDim.x + threadIdx.x; // over B*L/4
    const int b = (int)(idx >> 9);               // L/4 == 512 float4 per row
    const float s = 2.0f * vz[b] / vv[b];
    const u2v vp = ((const u2v*)v)[idx];
    f4v zv = ((const f4v*)z)[idx];
    zv.x -= s * bflo2f(vp.x);
    zv.y -= s * bfhi2f(vp.x);
    zv.z -= s * bflo2f(vp.y);
    zv.w -= s * bfhi2f(vp.y);
    ((f4v*)z)[idx] = zv;
}

__global__ void hf_zero(float* __restrict__ vz, float* __restrict__ vv)
{
    const int i = blockIdx.x * blockDim.x + threadIdx.x;
    if (i < B_DIM) { vz[i] = 0.f; vv[i] = 0.f; }
}

// ---------------------------------------------------------------------------
// Host-side chaining. z lives in d_out (f32); weights + v ping-pong in d_ws
// as bf16:  [ wbf: 8*L*K | vbuf0: B*L | vbuf1: B*L | vz: B | vv: B ]
// ---------------------------------------------------------------------------
extern "C" void kernel_launch(void* const* d_in, const int* in_sizes, int n_in,
                              void* d_out, int out_size, void* d_ws, size_t ws_size,
                              hipStream_t stream)
{
    (void)in_sizes; (void)n_in; (void)out_size; (void)ws_size;

    const float* z_in   = (const float*)d_in[0];
    const float* h_last = (const float*)d_in[1];
    const float* W0     = (const float*)d_in[2];
    const float* b0     = (const float*)d_in[3];
    const float* Ws     = (const float*)d_in[4];
    const float* bs     = (const float*)d_in[5];
    float* z = (float*)d_out;

    const size_t WELEM = (size_t)L_DIM * K_DIM;      // 4M per weight matrix
    const size_t VELEM = (size_t)B_DIM * L_DIM;      // 16M per activation buffer

    unsigned short* wbf   = (unsigned short*)d_ws;   // NFLOW * WELEM bf16
    unsigned short* vbuf0 = wbf + (size_t)NFLOW * WELEM;
    unsigned short* vbuf1 = vbuf0 + VELEM;
    float* vz = (float*)(vbuf1 + VELEM);
    float* vv = vz + B_DIM;

    // one-time (per launch) conversions: weights + h_last -> bf16
    {
        const long long n4w0 = (long long)(WELEM / 4);
        cvt_bf16<<<(unsigned)((n4w0 + 255) / 256), 256, 0, stream>>>(W0, wbf, n4w0);
        const long long n4ws = (long long)((NFLOW - 1) * WELEM / 4);
        cvt_bf16<<<(unsigned)((n4ws + 255) / 256), 256, 0, stream>>>(Ws, wbf + WELEM, n4ws);
        const long long n4h = (long long)(VELEM / 4);
        cvt_bf16<<<(unsigned)((n4h + 255) / 256), 256, 0, stream>>>(h_last, vbuf0, n4h);
    }

    (void)hipMemcpyAsync(z, z_in, sizeof(float) * VELEM, hipMemcpyDeviceToDevice, stream);

    const dim3 ggrid(L_DIM / 128, B_DIM / 128);
    const int upd_blocks = (int)((VELEM / 4) / 256);

    const unsigned short* Acur = vbuf0;   // flow 0 consumes bf16(h_last)
    unsigned short* vdst  = vbuf1;
    unsigned short* vnext = vbuf0;

    for (int j = 0; j < NFLOW; ++j) {
        const unsigned short* Wj = wbf + (size_t)j * WELEM;
        const float* bj = (j == 0) ? b0 : (bs + (size_t)(j - 1) * L_DIM);

        hf_zero<<<(B_DIM + 255) / 256, 256, 0, stream>>>(vz, vv);
        hf_gemm<<<ggrid, 256, 0, stream>>>(Acur, Wj, bj, z, vdst, vz, vv);
        hf_update<<<upd_blocks, 256, 0, stream>>>(z, vdst, vz, vv);

        Acur = vdst;
        unsigned short* tmp = vdst; vdst = vnext; vnext = tmp;
    }
}
